// PiProducts_20985210208769
// MI455X (gfx1250) — 8-bit matrix operand order repaired and checked by reading, compiled, not run on hardware
//
#include <hip/hip_runtime.h>
#include <hip/hip_bf16.h>
#include <stdint.h>

typedef __attribute__((ext_vector_type(16))) int   v16i;
typedef __attribute__((ext_vector_type(8)))  float v8f;

#define GG   512
#define TT   128
#define PP   256
#define BB   512

__global__ void cvt_fp8_kernel(const float* __restrict__ in,
                               uint32_t* __restrict__ out, int n4) {
    int i = blockIdx.x * blockDim.x + threadIdx.x;
    if (i >= n4) return;
    float4 f = ((const float4*)in)[i];
    uint32_t w = (f.x != 0.0f ? 0x38u : 0u)
               | ((f.y != 0.0f ? 0x38u : 0u) << 8)
               | ((f.z != 0.0f ? 0x38u : 0u) << 16)
               | ((f.w != 0.0f ? 0x38u : 0u) << 24);
    out[i] = w;
}

__global__ __launch_bounds__(32)
__attribute__((amdgpu_waves_per_eu(4)))
void gf2_wmma_kernel(const uint8_t* __restrict__ Apsi,
                     const uint8_t* __restrict__ Aphi,
                     const uint8_t* __restrict__ Bprm,
                     const int* __restrict__ psiC,
                     const int* __restrict__ phiC,
                     float* __restrict__ out) {
    const uint32_t lane = threadIdx.x;
    const uint32_t col  = lane & 15u;
    const uint32_t hi   = lane >> 4;
    const uint32_t g      = blockIdx.x;
    const uint32_t bchunk = blockIdx.y;
    const uint32_t bbase  = bchunk * 32u;

    v16i Bf[2][2];
#pragma unroll
    for (int bt = 0; bt < 2; ++bt) {
        const uint32_t rowoff = (bbase + (uint32_t)bt * 16u + col) * (uint32_t)PP
                              + hi * 16u;
#pragma unroll
        for (int kh = 0; kh < 2; ++kh) {
            v16i bf;
#pragma unroll
            for (int q = 0; q < 4; ++q) {
                int4 d = *(const int4*)(Bprm + rowoff + (uint32_t)(kh * 128 + 32 * q));
                bf[4 * q + 0] = d.x; bf[4 * q + 1] = d.y;
                bf[4 * q + 2] = d.z; bf[4 * q + 3] = d.w;
            }
            Bf[bt][kh] = bf;
        }
    }

    const uint32_t aoff = (g * (uint32_t)TT + col) * (uint32_t)PP + hi * 16u;
    const uint32_t coff = g * (uint32_t)TT + hi * 8u;

    uint32_t acc[2] = {0u, 0u};

#pragma unroll
    for (uint32_t tile = 0; tile < 8; ++tile) {
        const uint32_t abase = aoff + tile * 16u * (uint32_t)PP;
        const uint32_t cbase = coff + tile * 16u;

        if (tile < 7) __builtin_prefetch(Apsi + abase + 16u * PP, 0, 1);
        v16i A0, A1;
#pragma unroll
        for (int j = 0; j < 8; ++j) {
            int2 d0 = *(const int2*)(Apsi + abase + (uint32_t)(32 * (j >> 1) + 8 * (j & 1)));
            int2 d1 = *(const int2*)(Apsi + abase + (uint32_t)(128 + 32 * (j >> 1) + 8 * (j & 1)));
            A0[2 * j] = d0.x; A0[2 * j + 1] = d0.y;
            A1[2 * j] = d1.x; A1[2 * j + 1] = d1.y;
        }
        int4 c0 = *(const int4*)(psiC + cbase);
        int4 c1 = *(const int4*)(psiC + cbase + 4);
        v8f Cp = {(float)c0.x, (float)c0.y, (float)c0.z, (float)c0.w,
                  (float)c1.x, (float)c1.y, (float)c1.z, (float)c1.w};

        uint32_t ppack[2];
#pragma unroll
        for (int bt = 0; bt < 2; ++bt) {
            v8f D = __builtin_amdgcn_wmma_f32_16x16x128_fp8_fp8(
                A0, Bf[bt][0], (short)0, Cp, false, false);
            D = __builtin_amdgcn_wmma_f32_16x16x128_fp8_fp8(
                A1, Bf[bt][1], (short)0, D, false, false);
            uint32_t pk = 0;
#pragma unroll
            for (int r = 0; r < 8; ++r)
                pk |= (uint32_t)((int)D[r] & 1) << r;
            ppack[bt] = pk;
        }

        if (tile < 7) __builtin_prefetch(Aphi + abase + 16u * PP, 0, 1);
#pragma unroll
        for (int j = 0; j < 8; ++j) {
            int2 d0 = *(const int2*)(Aphi + abase + (uint32_t)(32 * (j >> 1) + 8 * (j & 1)));
            int2 d1 = *(const int2*)(Aphi + abase + (uint32_t)(128 + 32 * (j >> 1) + 8 * (j & 1)));
            A0[2 * j] = d0.x; A0[2 * j + 1] = d0.y;
            A1[2 * j] = d1.x; A1[2 * j + 1] = d1.y;
        }
        c0 = *(const int4*)(phiC + cbase);
        c1 = *(const int4*)(phiC + cbase + 4);
        v8f Cq = {(float)c0.x, (float)c0.y, (float)c0.z, (float)c0.w,
                  (float)c1.x, (float)c1.y, (float)c1.z, (float)c1.w};

#pragma unroll
        for (int bt = 0; bt < 2; ++bt) {
            v8f D = __builtin_amdgcn_wmma_f32_16x16x128_fp8_fp8(
                A0, Bf[bt][0], (short)0, Cq, false, false);
            D = __builtin_amdgcn_wmma_f32_16x16x128_fp8_fp8(
                A1, Bf[bt][1], (short)0, D, false, false);
            uint32_t qk = 0;
#pragma unroll
            for (int r = 0; r < 8; ++r)
                qk |= (uint32_t)((int)D[r] & 1) << r;
            acc[bt] ^= (uint32_t)__popc(ppack[bt] & qk) & 1u;
        }
    }

#pragma unroll
    for (int bt = 0; bt < 2; ++bt) {
        uint32_t par = (acc[bt] ^ (uint32_t)__shfl_xor((int)acc[bt], 16, 32)) & 1u;
        if (lane < 16) {
            uint32_t b = bbase + (uint32_t)bt * 16u + lane;
            float s = 1.0f - 2.0f * (float)par;
            float4 o = {s, 0.0f, 0.0f, 0.0f};
            *(float4*)(out + (b * (uint32_t)GG + g) * 4u) = o;
        }
    }
}

extern "C" void kernel_launch(void* const* d_in, const int* in_sizes, int n_in,
                              void* d_out, int out_size, void* d_ws, size_t ws_size,
                              hipStream_t stream) {
    (void)in_sizes; (void)n_in; (void)out_size; (void)ws_size;
    const int*   psiC = (const int*)d_in[0];
    const float* psiP = (const float*)d_in[1];
    const int*   phiC = (const int*)d_in[2];
    const float* phiP = (const float*)d_in[3];
    const float* prm  = (const float*)d_in[4];

    uint8_t* ws   = (uint8_t*)d_ws;
    uint8_t* psi8 = ws;
    uint8_t* phi8 = ws + (size_t)GG * TT * PP;
    uint8_t* prm8 = ws + 2 * (size_t)GG * TT * PP;

    const int n4 = GG * TT * PP / 4;
    const int m4 = BB * PP / 4;
    cvt_fp8_kernel<<<(n4 + 255) / 256, 256, 0, stream>>>(psiP, (uint32_t*)psi8, n4);
    cvt_fp8_kernel<<<(n4 + 255) / 256, 256, 0, stream>>>(phiP, (uint32_t*)phi8, n4);
    cvt_fp8_kernel<<<(m4 + 255) / 256, 256, 0, stream>>>(prm,  (uint32_t*)prm8, m4);

    dim3 grid(GG, BB / 32);
    gf2_wmma_kernel<<<grid, 32, 0, stream>>>(psi8, phi8, prm8, psiC, phiC,
                                             (float*)d_out);
}
